// DistKernelRegressor_8126078124619
// MI455X (gfx1250) — matrix operand order repaired and checked by reading, not compiled, not run on hardware
//
#include <hip/hip_runtime.h>
#include <hip/hip_bf16.h>

typedef __attribute__((ext_vector_type(16))) _Float16 v16h;
typedef __attribute__((ext_vector_type(8)))  float    v8f;

#define N_NODES 1024
#define H_DIM   128
#define K_DIM   64          // H_DIM / 2
#define NEG_SLOPE 0.2f

#define TILE 16
#define WAVES_PER_BLOCK 16
#define BLOCK_THREADS (WAVES_PER_BLOCK * 32)

// LDS row paddings to break 64-bank conflict patterns
#define HPAD 132            // floats per h-row
#define WPAD 136            // halfs per W1 row

// |x| as a single VOP3 max with a negate source-modifier.
// Kept in inline asm so InstCombine cannot re-canonicalize it to llvm.fabs:
// vectorized (v2f32) fabs triggers an isel crash in the gfx1250 backend.
__device__ __forceinline__ float abs_asm(float x) {
    float r;
    asm("v_max_num_f32_e64 %0, %1, -%1" : "=v"(r) : "v"(x));
    return r;
}

__global__ __launch_bounds__(BLOCK_THREADS)
void distreg_pair_kernel(const float* __restrict__ hiddens,
                         const float* __restrict__ fc1_w,
                         const float* __restrict__ fc1_b,
                         const float* __restrict__ fc2_w,
                         const float* __restrict__ fc2_b,
                         float* __restrict__ out_kernel)
{
    __shared__ float    sHI[TILE][HPAD];      // 16 i-rows of hiddens (f32)
    __shared__ float    sHJ[TILE][HPAD];      // 16 j-rows of hiddens (f32)
    __shared__ _Float16 sW1[K_DIM][WPAD];     // fc1_w as f16, row-major [k][h]

    const int it  = blockIdx.y;               // i tile
    const int jt  = blockIdx.x;               // j tile
    const int tid = threadIdx.x;

    // ---- cooperative LDS fill (coalesced) ----
    for (int idx = tid; idx < TILE * H_DIM; idx += BLOCK_THREADS) {
        const int r = idx >> 7, c = idx & 127;
        sHI[r][c] = hiddens[(it * TILE + r) * H_DIM + c];
        sHJ[r][c] = hiddens[(jt * TILE + r) * H_DIM + c];
    }
    for (int idx = tid; idx < K_DIM * H_DIM; idx += BLOCK_THREADS) {
        const int r = idx >> 7, c = idx & 127;
        sW1[r][c] = (_Float16)fc1_w[idx];
    }
    __syncthreads();

    const int wave = tid >> 5;                // i_local: one wave per i-row
    const int lane = tid & 31;
    const int m    = lane & 15;               // A: M = j_local ; B/D: N = k_local
    const int hiA  = (lane >> 4) * 8;         // A fragment K sub-offset per half-wave
    const int hiB  = (lane >> 4) * 8;         // B fragment K sub-offset per half-wave

    v8f acc[4];
    #pragma unroll
    for (int kt = 0; kt < 4; ++kt) acc[kt] = v8f{};

    const float* hi_row = &sHI[wave][0];      // broadcast read within wave
    const float* hj_row = &sHJ[m][0];

    // ---- main WMMA loop: z[16 j x 64 k] = |h_i - h_j| @ W1^T ----
    #pragma unroll
    for (int hs = 0; hs < 4; ++hs) {          // K (=h) in chunks of 32
        // A fragment: 16x32 f16, CDNA5 layout:
        //   lanes 0-15:  M=lane,    elem e -> K = (e>>3)*16 + (e&7)
        //   lanes 16-31: M=lane-16, elem e -> K = (e>>3)*16 + 8 + (e&7)
        v16h a;
        #pragma unroll
        for (int g = 0; g < 2; ++g) {
            const int base = hs * 32 + g * 16 + hiA;
            #pragma unroll
            for (int w = 0; w < 8; ++w) {
                const float d = abs_asm(hi_row[base + w] - hj_row[base + w]);
                a[g * 8 + w] = (_Float16)d;
            }
        }
        #pragma unroll
        for (int kt = 0; kt < 4; ++kt) {      // N (=k) in tiles of 16
            // B fragment: 32x16 f16; same K layout as A:
            // elem e -> K = (e>>3)*16 + 8*(lane>>4) + (e&7).
            v16h b;
            const _Float16* wrow = &sW1[kt * 16 + m][hs * 32 + hiB];
            #pragma unroll
            for (int e = 0; e < 16; ++e) b[e] = wrow[(e >> 3) * 16 + (e & 7)];

            acc[kt] = __builtin_amdgcn_wmma_f32_16x16x32_f16(
                /*neg_a=*/false, a, /*neg_b=*/false, b,
                /*c_mod=*/(short)0, acc[kt],
                /*reuse_a=*/false, /*reuse_b=*/false);
        }
    }

    // ---- epilogue: +b1, LeakyReLU, dot with w2, reduce over k, sigmoid ----
    float b1v[4], w2v[4];
    #pragma unroll
    for (int kt = 0; kt < 4; ++kt) {
        b1v[kt] = fc1_b[kt * 16 + m];         // k = kt*16 + (lane&15)
        w2v[kt] = fc2_w[kt * 16 + m];
    }
    const float b2 = fc2_b[0];
    const int  i   = it * TILE + wave;

    #pragma unroll
    for (int r = 0; r < 8; ++r) {             // D layout: VGPR r -> j = r (+8 in hi half)
        float t = 0.f;
        #pragma unroll
        for (int kt = 0; kt < 4; ++kt) {
            float z = acc[kt][r] + b1v[kt];
            z = (z >= 0.f) ? z : NEG_SLOPE * z;
            t += z * w2v[kt];
        }
        // reduce the 16 k-lanes of each half-wave
        t += __shfl_xor(t, 1, 32);
        t += __shfl_xor(t, 2, 32);
        t += __shfl_xor(t, 4, 32);
        t += __shfl_xor(t, 8, 32);

        const float e    = t + b2;
        const float sval = 1.0f / (1.0f + __expf(-e));
        const int   j    = jt * TILE + r + ((lane >> 4) << 3);
        if (m == 0) out_kernel[i * N_NODES + j] = sval;
    }
}

extern "C" void kernel_launch(void* const* d_in, const int* in_sizes, int n_in,
                              void* d_out, int out_size, void* d_ws, size_t ws_size,
                              hipStream_t stream) {
    const float* hiddens = (const float*)d_in[0];
    const float* fc1_w   = (const float*)d_in[1];
    const float* fc1_b   = (const float*)d_in[2];
    const float* fc2_w   = (const float*)d_in[3];
    const float* fc2_b   = (const float*)d_in[4];

    float* out = (float*)d_out;

    // Output tuple = (hiddens, kernel) concatenated flat.
    (void)hipMemcpyAsync(out, hiddens, (size_t)N_NODES * H_DIM * sizeof(float),
                         hipMemcpyDeviceToDevice, stream);
    float* out_kernel = out + (size_t)N_NODES * H_DIM;

    dim3 grid(N_NODES / TILE, N_NODES / TILE);   // 64 x 64 blocks
    distreg_pair_kernel<<<grid, BLOCK_THREADS, 0, stream>>>(
        hiddens, fc1_w, fc1_b, fc2_w, fc2_b, out_kernel);
}
